// GraphStateEncoder_30331059044746
// MI455X (gfx1250) — matrix operand order repaired and checked by reading, not compiled, not run on hardware
//
#include <hip/hip_runtime.h>

#define HID 128
#define ROLE_D 16
#define IN_DIM 28
#define XPAD 40    // bf16 row stride for 32-wide tiles (multiple of 8, bank-spread)
#define TPAD 136   // bf16 row stride for 128-wide tiles (multiple of 8, bank-spread)
#define ZPAD 132   // f32 row stride for LayerNorm tile

typedef __attribute__((ext_vector_type(16))) __bf16 v16bf;
typedef __attribute__((ext_vector_type(8)))  __bf16 v8bf;
typedef __attribute__((ext_vector_type(8)))  float  v8f;

static __device__ __forceinline__ __bf16 f2bf(float f) {
  unsigned u = __builtin_bit_cast(unsigned, f);
  u = (u + 0x7FFFu + ((u >> 16) & 1u)) >> 16;
  unsigned short s = (unsigned short)u;
  return __builtin_bit_cast(__bf16, s);
}

// A fragment (16x32 bf16) from a bf16 LDS tile, row stride sx (elements, mult of 8).
// ISA layout: lane -> row M = lane&15; half = lane>>4;
// vgprs 0..3 hold K = half*8 + 0..7 (pairs), vgprs 4..7 hold K = 16+half*8 + 0..7.
// Both spans are contiguous 16B in a bf16 row-major tile -> 2x ds_load_b128.
static __device__ __forceinline__ v16bf load_a_bf(const __bf16* X, int sx, int lane) {
  int r = lane & 15, half = lane >> 4;
  const __bf16* p = X + r * sx + half * 8;
  v8bf lo = *(const v8bf*)p;
  v8bf hi = *(const v8bf*)(p + 16);
  return __builtin_shufflevector(lo, hi, 0, 1, 2, 3, 4, 5, 6, 7,
                                 8, 9, 10, 11, 12, 13, 14, 15);
}

// B fragment (32x16 bf16) from pre-converted bf16 weights W[k][n] (ldw = 128).
// ISA layout: lane -> col N = lane&15; elems 0..7 K = half*8+0..7, elems 8..15 K = 16+half*8+0..7.
static __device__ __forceinline__ v16bf load_b_bf(const __bf16* W, int colbase,
                                                  int k0, int lane) {
  int n = lane & 15, half = lane >> 4;
  const __bf16* p = W + (long)(k0 + half * 8) * HID + colbase + n;
  v16bf b;
#pragma unroll
  for (int s = 0; s < 8; ++s) {
    b[s] = p[s * HID];
    b[s + 8] = p[(16 + s) * HID];
  }
  return b;
}

__global__ void k_zero4(float4* p, long count4) {
  long i = (long)blockIdx.x * blockDim.x + threadIdx.x;
  long stride = (long)gridDim.x * blockDim.x;
  float4 z = make_float4(0.f, 0.f, 0.f, 0.f);
  for (; i < count4; i += stride) p[i] = z;
}

// One-shot weight conversion f32 -> bf16 (W1 zero-padded to 32 rows).
__global__ void k_cvt_w(const float* __restrict__ W1, const float* __restrict__ W2,
                        const float* __restrict__ M1, const float* __restrict__ M2,
                        __bf16* __restrict__ w1b, __bf16* __restrict__ w2b,
                        __bf16* __restrict__ m1b, __bf16* __restrict__ m2b) {
  int i = blockIdx.x * blockDim.x + threadIdx.x;
  if (i < 32 * HID) w1b[i] = f2bf(i < IN_DIM * HID ? W1[i] : 0.0f);
  if (i < HID * HID) {
    w2b[i] = f2bf(W2[i]);
    m1b[i] = f2bf(M1[i]);
    m2b[i] = f2bf(M2[i]);
  }
}

__global__ void k_deg(const int* __restrict__ src, const int* __restrict__ dst,
                      const float* __restrict__ bw, const int* __restrict__ pblk,
                      const float* __restrict__ pw, float* __restrict__ deg,
                      int EB, int EP) {
  int i = blockIdx.x * blockDim.x + threadIdx.x;
  if (i < EB) {
    float w = bw[i];
    unsafeAtomicAdd(deg + src[i], w);
    unsafeAtomicAdd(deg + dst[i], w);
  } else if (i < EB + EP) {
    int j = i - EB;
    unsafeAtomicAdd(deg + pblk[j], pw[j]);
  }
}

// h = relu(relu(x@W1+b1)@W2+b2). 16 rows/block, 8 waves = 8 column tiles of 16.
__global__ __launch_bounds__(256)
void k_input_mlp(const float* __restrict__ area, const float* __restrict__ cons,
                 const float* __restrict__ place, const int* __restrict__ role_ids,
                 const float* __restrict__ role_emb, const float* __restrict__ deg,
                 const __bf16* __restrict__ w1b, const float* __restrict__ b1v,
                 const __bf16* __restrict__ w2b, const float* __restrict__ b2v,
                 float* __restrict__ h, int N) {
  __shared__ __align__(16) __bf16 xs[16][XPAD];
  __shared__ __align__(16) __bf16 h1[16][TPAD];
  int tid = threadIdx.x;
  int row0 = blockIdx.x * 16;
#pragma unroll
  for (int t = 0; t < 2; ++t) {
    int e = t * 256 + tid;                 // 512 entries = 16 rows x 32 cols
    int r = e >> 5, c = e & 31;
    int g = row0 + r; if (g >= N) g = N - 1;
    float v;
    if (c == 0)        v = area[g];
    else if (c <= 5)   v = cons[g * 5 + (c - 1)];
    else if (c == 6)   v = deg[g];
    else if (c <= 11)  v = place[g * 5 + (c - 7)];
    else if (c <= 27)  v = role_emb[role_ids[g] * ROLE_D + (c - 12)];
    else               v = 0.0f;           // K pad 28..31
    xs[r][c] = f2bf(v);
  }
  __syncthreads();
  int wave = tid >> 5, lane = tid & 31;
  int colbase = wave * 16, n = lane & 15, half = lane >> 4;
  { // layer 1: K=32 (padded) -> single WMMA
    v16bf a  = load_a_bf(&xs[0][0], XPAD, lane);
    v16bf bm = load_b_bf(w1b, colbase, 0, lane);
    v8f c; float bias = b1v[colbase + n];
#pragma unroll
    for (int i = 0; i < 8; ++i) c[i] = bias;
    c = __builtin_amdgcn_wmma_f32_16x16x32_bf16(false, a, false, bm, (short)0, c, false, false);
#pragma unroll
    for (int i = 0; i < 8; ++i) h1[i + 8 * half][colbase + n] = f2bf(fmaxf(c[i], 0.0f));
  }
  __syncthreads();
  { // layer 2: K=128 -> 4 chained WMMAs
    v8f c; float bias = b2v[colbase + n];
#pragma unroll
    for (int i = 0; i < 8; ++i) c[i] = bias;
#pragma unroll
    for (int k0 = 0; k0 < HID; k0 += 32) {
      v16bf a  = load_a_bf(&h1[0][k0], TPAD, lane);
      v16bf bm = load_b_bf(w2b, colbase, k0, lane);
      c = __builtin_amdgcn_wmma_f32_16x16x32_bf16(false, a, false, bm, (short)0, c, false, false);
    }
#pragma unroll
    for (int i = 0; i < 8; ++i) {
      int g = row0 + i + 8 * half;
      if (g < N) h[(long)g * HID + colbase + n] = fmaxf(c[i], 0.0f);
    }
  }
}

// One wave per edge; each lane owns 4 contiguous floats (float4 gather, 4 f32 atomics).
// h (51MB) and agg (51MB) both fit in the 192MB L2 -> atomics run at L2 speed.
__global__ __launch_bounds__(256)
void k_edge_agg(const int* __restrict__ src, const int* __restrict__ dst,
                const float* __restrict__ bw, const int* __restrict__ pblk,
                const float* __restrict__ pw, const float* __restrict__ h,
                float* __restrict__ agg, int EB, int EP) {
  int wave = threadIdx.x >> 5, lane = threadIdx.x & 31;
  long wid = (long)blockIdx.x * 8 + wave;
  int base = lane * 4;
  if (wid < EB) {
    int e = (int)wid;
    int s = src[e], d = dst[e];
    float w = bw[e];
    float4 hd = *(const float4*)(h + (long)d * HID + base);
    float4 hs = *(const float4*)(h + (long)s * HID + base);
    float* as = agg + (long)s * HID + base;
    float* ad = agg + (long)d * HID + base;
    unsafeAtomicAdd(as + 0, w * hd.x); unsafeAtomicAdd(as + 1, w * hd.y);
    unsafeAtomicAdd(as + 2, w * hd.z); unsafeAtomicAdd(as + 3, w * hd.w);
    unsafeAtomicAdd(ad + 0, w * hs.x); unsafeAtomicAdd(ad + 1, w * hs.y);
    unsafeAtomicAdd(ad + 2, w * hs.z); unsafeAtomicAdd(ad + 3, w * hs.w);
  } else if (wid < (long)EB + EP) {
    int e = (int)(wid - EB);
    int bk = pblk[e];
    float w = pw[e];
    float4 hb = *(const float4*)(h + (long)bk * HID + base);
    float* ab = agg + (long)bk * HID + base;
    unsafeAtomicAdd(ab + 0, w * hb.x); unsafeAtomicAdd(ab + 1, w * hb.y);
    unsafeAtomicAdd(ab + 2, w * hb.z); unsafeAtomicAdd(ab + 3, w * hb.w);
  }
}

// m = relu(agg@M1+mb1)@M2+mb2; z=h+m; LayerNorm; emb, mask, column sums for graph mean.
__global__ __launch_bounds__(256)
void k_msg_ln(const float* __restrict__ agg, const float* __restrict__ h,
              const __bf16* __restrict__ m1b, const float* __restrict__ mb1,
              const __bf16* __restrict__ m2b, const float* __restrict__ mb2,
              const float* __restrict__ gam, const float* __restrict__ bet,
              float* __restrict__ emb, float* __restrict__ mask,
              float* __restrict__ gsum, int N) {
  __shared__ __align__(16) __bf16 as_[16][TPAD];
  __shared__ __align__(16) __bf16 hm[16][TPAD];
  __shared__ float zb[16][ZPAD];
  __shared__ float muv[16], rsv[16];
  __shared__ float colp[2][HID];
  int tid = threadIdx.x;
  int row0 = blockIdx.x * 16;
#pragma unroll
  for (int i = 0; i < 8; ++i) {          // coalesced stage of agg tile -> bf16
    int e = i * 256 + tid;
    int r = e >> 7, c = e & 127;
    int g = row0 + r; if (g >= N) g = N - 1;
    as_[r][c] = f2bf(agg[(long)g * HID + c]);
  }
  __syncthreads();
  int wave = tid >> 5, lane = tid & 31;
  int colbase = wave * 16, n = lane & 15, half = lane >> 4;
  { // GEMM1 + relu
    v8f c; float bias = mb1[colbase + n];
#pragma unroll
    for (int i = 0; i < 8; ++i) c[i] = bias;
#pragma unroll
    for (int k0 = 0; k0 < HID; k0 += 32) {
      v16bf a  = load_a_bf(&as_[0][k0], TPAD, lane);
      v16bf bm = load_b_bf(m1b, colbase, k0, lane);
      c = __builtin_amdgcn_wmma_f32_16x16x32_bf16(false, a, false, bm, (short)0, c, false, false);
    }
#pragma unroll
    for (int i = 0; i < 8; ++i) hm[i + 8 * half][colbase + n] = f2bf(fmaxf(c[i], 0.0f));
  }
  __syncthreads();
  { // GEMM2, then z = h + m
    v8f c; float bias = mb2[colbase + n];
#pragma unroll
    for (int i = 0; i < 8; ++i) c[i] = bias;
#pragma unroll
    for (int k0 = 0; k0 < HID; k0 += 32) {
      v16bf a  = load_a_bf(&hm[0][k0], TPAD, lane);
      v16bf bm = load_b_bf(m2b, colbase, k0, lane);
      c = __builtin_amdgcn_wmma_f32_16x16x32_bf16(false, a, false, bm, (short)0, c, false, false);
    }
#pragma unroll
    for (int i = 0; i < 8; ++i) {
      int m = i + 8 * half;
      int g = row0 + m; if (g >= N) g = N - 1;
      zb[m][colbase + n] = c[i] + h[(long)g * HID + colbase + n];
    }
  }
  __syncthreads();
  if (tid < 16) {                         // per-row mean / rstd
    float s = 0.0f, s2 = 0.0f;
#pragma unroll
    for (int c = 0; c < HID; ++c) { float z = zb[tid][c]; s += z; s2 += z * z; }
    float mu = s * (1.0f / HID);
    float var = s2 * (1.0f / HID) - mu * mu;
    muv[tid] = mu;
    rsv[tid] = rsqrtf(var + 1e-5f);
  }
  __syncthreads();
  { // normalize + write + per-column partials for graph mean
    int c = tid & 127, rh = tid >> 7;
    float gg = gam[c], bb = bet[c];
    float cs = 0.0f;
#pragma unroll
    for (int i = 0; i < 8; ++i) {
      int m = rh * 8 + i;
      int gr = row0 + m;
      float v = (zb[m][c] - muv[m]) * rsv[m] * gg + bb;
      if (gr < N) { emb[(long)gr * HID + c] = v; cs += v; }
    }
    colp[rh][c] = cs;
  }
  if (tid < 16 && row0 + tid < N) mask[row0 + tid] = 1.0f;
  __syncthreads();
  if (tid < HID) unsafeAtomicAdd(gsum + tid, colp[0][tid] + colp[1][tid]);
}

__global__ void k_fin(const float* __restrict__ gsum, float* __restrict__ gemb, int N) {
  int c = threadIdx.x;
  if (c < HID) gemb[c] = gsum[c] / (float)N;
}

extern "C" void kernel_launch(void* const* d_in, const int* in_sizes, int n_in,
                              void* d_out, int out_size, void* d_ws, size_t ws_size,
                              hipStream_t stream) {
  const float* area     = (const float*)d_in[0];
  const float* cons     = (const float*)d_in[1];
  const float* place    = (const float*)d_in[2];
  const int*   role_ids = (const int*)  d_in[3];
  const int*   b2b_src  = (const int*)  d_in[4];
  const int*   b2b_dst  = (const int*)  d_in[5];
  const float* b2b_w    = (const float*)d_in[6];
  const int*   p2b_blk  = (const int*)  d_in[7];
  const float* p2b_w    = (const float*)d_in[8];
  const float* role_emb = (const float*)d_in[9];
  const float* W1  = (const float*)d_in[10];
  const float* b1  = (const float*)d_in[11];
  const float* W2  = (const float*)d_in[12];
  const float* b2  = (const float*)d_in[13];
  const float* M1  = (const float*)d_in[14];
  const float* mb1 = (const float*)d_in[15];
  const float* M2  = (const float*)d_in[16];
  const float* mb2 = (const float*)d_in[17];
  const float* gam = (const float*)d_in[18];
  const float* bet = (const float*)d_in[19];

  int N  = in_sizes[0];
  int EB = in_sizes[4];
  int EP = in_sizes[7];

  float* ws   = (float*)d_ws;
  float* deg  = ws;                              // [N] f32
  float* h    = ws + N;                          // [N*128] f32
  float* agg  = h + (long)N * HID;               // [N*128] f32
  float* gsum = agg + (long)N * HID;             // [128] f32
  __bf16* w1b = (__bf16*)(gsum + HID);           // [32*128] bf16 (padded)
  __bf16* w2b = w1b + 32 * HID;                  // [128*128] bf16
  __bf16* m1b = w2b + HID * HID;                 // [128*128] bf16
  __bf16* m2b = m1b + HID * HID;                 // [128*128] bf16

  float* emb  = (float*)d_out;                   // [N*128]
  float* gemb = emb + (long)N * HID;             // [128]
  float* mask = gemb + HID;                      // [N]

  // zero deg, and agg..gsum (contiguous); h is fully overwritten, skip it
  k_zero4<<<128, 256, 0, stream>>>((float4*)deg, (long)N / 4);
  k_zero4<<<2048, 256, 0, stream>>>((float4*)agg, ((long)N * HID + HID) / 4);
  k_cvt_w<<<(HID * HID + 255) / 256, 256, 0, stream>>>(W1, W2, M1, M2, w1b, w2b, m1b, m2b);
  k_deg<<<(EB + EP + 255) / 256, 256, 0, stream>>>(b2b_src, b2b_dst, b2b_w,
                                                   p2b_blk, p2b_w, deg, EB, EP);
  int nb = (N + 15) / 16;
  k_input_mlp<<<nb, 256, 0, stream>>>(area, cons, place, role_ids, role_emb, deg,
                                      w1b, b1, w2b, b2, h, N);
  long waves = (long)EB + EP;
  k_edge_agg<<<(int)((waves + 7) / 8), 256, 0, stream>>>(b2b_src, b2b_dst, b2b_w,
                                                         p2b_blk, p2b_w, h, agg, EB, EP);
  k_msg_ln<<<nb, 256, 0, stream>>>(agg, h, m1b, mb1, m2b, mb2, gam, bet,
                                   emb, mask, gsum, N);
  k_fin<<<1, HID, 0, stream>>>(gsum, gemb, N);
}
